// Gate_50062138802428
// MI455X (gfx1250) — matrix operand order repaired and checked by reading, not compiled, not run on hardware
//
#include <hip/hip_runtime.h>
#include <hip/hip_bf16.h>

// ---- CDNA5 WMMA types ----
typedef __attribute__((ext_vector_type(16))) __bf16 v16bf;
typedef __attribute__((ext_vector_type(8)))  float  v8f;
typedef __attribute__((ext_vector_type(4)))  float  f32x4;

#define DIM     2048   // model dim (K)
#define NEXP    64     // routed experts (N of the gemm)
#define TOPK    6
#define TOKW    32     // tokens per wave (2 x 16-row WMMA tiles)
#define TPB     256    // tokens per block (8 waves x 32 tokens)
#define LSTRIDE 68     // LDS row stride in floats (64 + pad)

// ---------------------------------------------------------------------------
// Kernel 1: one-time f32 -> bf16 conversion of the gate weight into d_ws.
// ---------------------------------------------------------------------------
__global__ void gate_cvt_w_bf16(const float* __restrict__ w,
                                __bf16* __restrict__ wb, int n) {
    int i = blockIdx.x * blockDim.x + threadIdx.x;
    if (i < n) wb[i] = (__bf16)w[i];
}

// ---------------------------------------------------------------------------
// Load one 16x32 bf16 A fragment (contiguous K order, same as B) from an f32 row.
// lanes 0-15 hold K = k0+[0..15]; lanes 16-31: k0+[16..31].
// ---------------------------------------------------------------------------
__device__ __forceinline__ v16bf load_a_frag(const float* __restrict__ xrow,
                                             int kA) {
    f32x4 a0 = *(const f32x4*)(xrow + kA + 0);
    f32x4 a1 = *(const f32x4*)(xrow + kA + 4);
    f32x4 a2 = *(const f32x4*)(xrow + kA + 8);
    f32x4 a3 = *(const f32x4*)(xrow + kA + 12);
    v16bf a;
    a[0]  = (__bf16)a0[0]; a[1]  = (__bf16)a0[1];
    a[2]  = (__bf16)a0[2]; a[3]  = (__bf16)a0[3];
    a[4]  = (__bf16)a1[0]; a[5]  = (__bf16)a1[1];
    a[6]  = (__bf16)a1[2]; a[7]  = (__bf16)a1[3];
    a[8]  = (__bf16)a2[0]; a[9]  = (__bf16)a2[1];
    a[10] = (__bf16)a2[2]; a[11] = (__bf16)a2[3];
    a[12] = (__bf16)a3[0]; a[13] = (__bf16)a3[1];
    a[14] = (__bf16)a3[2]; a[15] = (__bf16)a3[3];
    return a;
}

// ---------------------------------------------------------------------------
// Kernel 2: fused gate. Each wave computes a 32-token x 64-expert logit tile
// (two 16-row WMMA tiles sharing every B fragment), then softmax + top-6
// per token out of LDS.
// ---------------------------------------------------------------------------
__global__ __launch_bounds__(256)
void gate_kernel(const float* __restrict__ x,
                 const __bf16* __restrict__ wb,
                 float* __restrict__ out_w,
                 float* __restrict__ out_i) {
    __shared__ float lds[TPB * LSTRIDE];

    const int tid  = threadIdx.x;
    const int wave = tid >> 5;        // 0..7 (wave32)
    const int lane = tid & 31;
    const int half = lane >> 4;       // lane half selects K sub-range
    const int lrow = lane & 15;       // token row (A) / expert col (B)
    const int tok0 = blockIdx.x * TPB + wave * TOKW;

    // acc[p][t]: token sub-tile p (rows 0-15 / 16-31), expert tile t (N 16t..)
    v8f acc00 = {}, acc01 = {}, acc02 = {}, acc03 = {};
    v8f acc10 = {}, acc11 = {}, acc12 = {}, acc13 = {};

    const float* xrow0 = x + (size_t)(tok0 + lrow) * DIM;
    const float* xrow1 = xrow0 + (size_t)16 * DIM;

    for (int k0 = 0; k0 < DIM; k0 += 32) {
        // prefetch next chunks of both streaming x rows (global_prefetch_b8)
        __builtin_prefetch(xrow0 + k0 + 64, 0, 1);
        __builtin_prefetch(xrow1 + k0 + 64, 0, 1);

        const int kA = k0 + half * 16;
        v16bf a0 = load_a_frag(xrow0, kA);
        v16bf a1 = load_a_frag(xrow1, kA);

        // ---- B fragments: 32x16 bf16 per expert tile; column N = lrow,
        // K contiguous (lanes 0-15: k0+[0..15], lanes 16-31: k0+[16..31]).
        // wb row-major [E][DIM] => B[k][n] = wb[n][k] = contiguous 32B.
        const int kB = k0 + half * 16;
        const __bf16* bbase = wb + (size_t)lrow * DIM + kB;
        v16bf b0 = *(const v16bf*)(bbase + 0 * 16 * DIM);
        v16bf b1 = *(const v16bf*)(bbase + 1 * 16 * DIM);
        v16bf b2 = *(const v16bf*)(bbase + 2 * 16 * DIM);
        v16bf b3 = *(const v16bf*)(bbase + 3 * 16 * DIM);

        // each B fragment feeds two independent WMMAs (two token sub-tiles)
        acc00 = __builtin_amdgcn_wmma_f32_16x16x32_bf16(false, a0, false, b0,
                                                        (short)0, acc00, false, false);
        acc10 = __builtin_amdgcn_wmma_f32_16x16x32_bf16(false, a1, false, b0,
                                                        (short)0, acc10, false, false);
        acc01 = __builtin_amdgcn_wmma_f32_16x16x32_bf16(false, a0, false, b1,
                                                        (short)0, acc01, false, false);
        acc11 = __builtin_amdgcn_wmma_f32_16x16x32_bf16(false, a1, false, b1,
                                                        (short)0, acc11, false, false);
        acc02 = __builtin_amdgcn_wmma_f32_16x16x32_bf16(false, a0, false, b2,
                                                        (short)0, acc02, false, false);
        acc12 = __builtin_amdgcn_wmma_f32_16x16x32_bf16(false, a1, false, b2,
                                                        (short)0, acc12, false, false);
        acc03 = __builtin_amdgcn_wmma_f32_16x16x32_bf16(false, a0, false, b3,
                                                        (short)0, acc03, false, false);
        acc13 = __builtin_amdgcn_wmma_f32_16x16x32_bf16(false, a1, false, b3,
                                                        (short)0, acc13, false, false);
    }

    // ---- scatter logits to LDS: C/D layout is VGPR j -> M=j (lanes 0-15)
    // or M=j+8 (lanes 16-31); N = lrow. lds[token][expert].
    {
        const int mb0 = wave * TOKW + (half ? 8 : 0);        // sub-tile 0 rows
        const int mb1 = mb0 + 16;                            // sub-tile 1 rows
        #pragma unroll
        for (int j = 0; j < 8; ++j) {
            float* r0 = &lds[(size_t)(mb0 + j) * LSTRIDE];
            r0[0 * 16 + lrow] = acc00[j];
            r0[1 * 16 + lrow] = acc01[j];
            r0[2 * 16 + lrow] = acc02[j];
            r0[3 * 16 + lrow] = acc03[j];
            float* r1 = &lds[(size_t)(mb1 + j) * LSTRIDE];
            r1[0 * 16 + lrow] = acc10[j];
            r1[1 * 16 + lrow] = acc11[j];
            r1[2 * 16 + lrow] = acc12[j];
            r1[3 * 16 + lrow] = acc13[j];
        }
    }
    __syncthreads();

    // ---- phase 2: per-token softmax + top-6 (256 tokens, 256 threads)
    {
        float* row = &lds[(size_t)tid * LSTRIDE];
        float m = -3.0e38f;
        #pragma unroll 8
        for (int i = 0; i < NEXP; ++i) m = fmaxf(m, row[i]);
        float s = 0.0f;
        #pragma unroll 8
        for (int i = 0; i < NEXP; ++i) {
            float p = __expf(row[i] - m);
            row[i] = p;
            s += p;
        }
        const float inv = 1.0f / s;   // ROUTE_SCALE == 1.0
        const size_t tok = (size_t)blockIdx.x * TPB + tid;
        #pragma unroll
        for (int j = 0; j < TOPK; ++j) {
            float best = -1.0f; int bi = 0;
            #pragma unroll 8
            for (int i = 0; i < NEXP; ++i) {
                float v = row[i];
                if (v > best) { best = v; bi = i; }
            }
            out_w[tok * TOPK + j] = best * inv;
            out_i[tok * TOPK + j] = (float)bi;
            row[bi] = -1.0f;
        }
    }
}

// ---------------------------------------------------------------------------
extern "C" void kernel_launch(void* const* d_in, const int* in_sizes, int n_in,
                              void* d_out, int out_size, void* d_ws, size_t ws_size,
                              hipStream_t stream) {
    const float* x = (const float*)d_in[0];   // [N, DIM] f32
    const float* w = (const float*)d_in[1];   // [E, DIM] f32
    float* out = (float*)d_out;               // [N*TOPK weights | N*TOPK indices]

    const int wn   = in_sizes[1];             // E * DIM elements
    const int ntok = in_sizes[0] / DIM;       // N tokens

    __bf16* wb = (__bf16*)d_ws;               // 256 KB bf16 weight copy

    gate_cvt_w_bf16<<<(wn + 255) / 256, 256, 0, stream>>>(w, wb, wn);

    float* out_w = out;
    float* out_i = out + (size_t)ntok * TOPK;
    gate_kernel<<<dim3(ntok / TPB), dim3(256), 0, stream>>>(x, wb, out_w, out_i);
}
